// LSTMModule_31808527794710
// MI455X (gfx1250) — matrix operand order repaired and checked by reading, not compiled, not run on hardware
//
#include <hip/hip_runtime.h>

typedef __attribute__((ext_vector_type(16))) __bf16 bf16x16;
typedef __attribute__((ext_vector_type(8)))  __bf16 bf16x8;
typedef __attribute__((ext_vector_type(8)))  float  f32x8;
typedef __attribute__((ext_vector_type(4)))  unsigned int u32x4;
typedef __attribute__((ext_vector_type(8)))  int i32x8;
typedef __attribute__((ext_vector_type(4)))  int i32x4;

#define B_   128
#define S_   512
#define F_   512
#define H_   1024
#define G4   4096      // 4*H
#define KTOT 1536      // F + H
#define NKT  48        // KTOT/32  (K-iterations per tile)
#define NKTX 16        // F/32     (K-iterations fed from x)
#define MT_  8         // B/16
#define NT_  64        // H/16
#define NBLK 128       // persistent grid size (2 blocks per nt)
#define GFRAG (NKT * 32 * 16)   // 24576 bf16 elems per (gate, nt) slice

// ---------------------------------------------------------------------------
// Pack [W_ih ; W_hh] (1536 x 4096 fp32) into bf16 WMMA B-fragment order:
//   wfrag[((ntg*NKT + kt)*32 + lane)*16 + e] = W[kt*32 + (lane>>4)*8 + (e&7) + (e>>3)*16][ntg*16 + (lane&15)]
// ---------------------------------------------------------------------------
__global__ void pack_w_kernel(const float* __restrict__ w_ih,
                              const float* __restrict__ w_hh,
                              __bf16* __restrict__ wfrag) {
    int tid = blockIdx.x * blockDim.x + threadIdx.x;      // 0 .. 4096*1536-1
    if (tid >= G4 * KTOT) return;
    int ntg  = tid / GFRAG;
    int rem  = tid % GFRAG;
    int kt   = rem / (32 * 16);
    int rem2 = rem % (32 * 16);
    int lane = rem2 / 16;
    int e    = rem2 % 16;
    int n = ntg * 16 + (lane & 15);
    int k = kt * 32 + (lane >> 4) * 8 + (e & 7) + (e >> 3) * 16;
    float v = (k < F_) ? w_ih[(size_t)k * G4 + n]
                       : w_hh[(size_t)(k - F_) * G4 + n];
    wfrag[tid] = (__bf16)v;
}

// fp32 -> bf16 copy of x (A-side operand feed)
__global__ void pack_x_kernel(const float* __restrict__ x,
                              __bf16* __restrict__ xbf, int n) {
    int i = blockIdx.x * blockDim.x + threadIdx.x;
    if (i < n) xbf[i] = (__bf16)x[i];
}

// zero h ping-pong buffers and the global barrier counter
__global__ void init_state_kernel(__bf16* __restrict__ h0,
                                  __bf16* __restrict__ h1,
                                  unsigned* __restrict__ barrier) {
    int i = blockIdx.x * blockDim.x + threadIdx.x;        // B_*H_ threads
    if (i < B_ * H_) {
        h0[i] = (__bf16)0.0f;
        h1[i] = (__bf16)0.0f;
    }
    if (i == 0) *barrier = 0u;
}

__device__ __forceinline__ float sigmoid_fast(float x) {
    return 1.0f / (1.0f + __expf(-x));
}

// ---------------------------------------------------------------------------
// Persistent LSTM kernel. 128 blocks x 4 waves = 512 waves, one per output
// tile group (mt, nt). Weights for this block's nt (all 4 gates, 192 KB) are
// DMA'd into LDS once by the Tensor Data Mover and stay resident for the
// whole sequence; c-state lives in VGPRs; h ping-pongs through L2; steps are
// separated by a software grid barrier.
// ---------------------------------------------------------------------------
__launch_bounds__(128, 1)
__global__ void lstm_persistent_kernel(const __bf16* __restrict__ xbf,
                                       const __bf16* __restrict__ wfrag,
                                       const float* __restrict__ bias,
                                       __bf16* __restrict__ h0,
                                       __bf16* __restrict__ h1,
                                       float* __restrict__ out,
                                       unsigned* __restrict__ barrier) {
    __shared__ __align__(128) __bf16 bsh[4 * GFRAG];      // 192 KB

    const int lane = threadIdx.x & 31;
    const int wave = threadIdx.x >> 5;
    const int nt   = blockIdx.x >> 1;                 // 0..63 (block pair shares nt)
    const int mt   = ((blockIdx.x & 1) << 2) + wave;  // 0..7
    const int hi   = lane >> 4;
    const int nl   = lane & 15;
    const int m    = mt * 16 + nl;                    // A row owned by this lane

    // ---- one-time: TDM DMA of this nt's weight slice (4 gates) into LDS ----
    // 2D tile: tile_dim0 = GFRAG elems (one gate row, 48 KB), tile_dim1 = 4
    // gate rows, row stride = NT_*GFRAG elems. LDS receives rows contiguously.
    if (wave == 0) {
        unsigned long long gaddr =
            (unsigned long long)(uintptr_t)(wfrag + (size_t)nt * GFRAG);
        unsigned ldsoff = (unsigned)(uintptr_t)(&bsh[0]);   // LDS byte offset

        u32x4 g0;
        g0.x = 1u;                                          // count=1 (user D#)
        g0.y = ldsoff;                                      // lds_addr
        g0.z = (unsigned)(gaddr & 0xffffffffu);             // global_addr[31:0]
        g0.w = (unsigned)((gaddr >> 32) & 0x1ffffffu)       // global_addr[56:32]
             | (2u << 30);                                  // type = 2 ("image")

        i32x8 g1;
        g1[0] = (int)(1u << 16);                            // data_size=2B, mask=0
        g1[1] = (int)((GFRAG & 0xffffu) << 16);             // tensor_dim0[15:0]
        g1[2] = (int)(((GFRAG >> 16) & 0xffffu)             // tensor_dim0[31:16]
                      | (4u << 16));                        // tensor_dim1 = 4
        g1[3] = (int)((GFRAG & 0xffffu) << 16);             // tile_dim0 = GFRAG
        g1[4] = 4;                                          // tile_dim1 = 4
        g1[5] = (int)(NT_ * GFRAG);                         // dim0_stride[31:0]
        g1[6] = 0;                                          // dim0_stride[47:32]
        g1[7] = 0;

        i32x4 z4 = {0, 0, 0, 0};
        i32x8 z8 = {0, 0, 0, 0, 0, 0, 0, 0};
        __builtin_amdgcn_tensor_load_to_lds(g0, g1, z4, z4, z8, 0);
        __builtin_amdgcn_s_wait_tensorcnt(0);
    }
    __syncthreads();

    // ---- loop-invariant state ----
    const int ng = nt * 16 + nl;                      // hidden index (C/D column)
    const float bi = bias[0 * H_ + ng];
    const float bf = bias[1 * H_ + ng];
    const float bg = bias[2 * H_ + ng];
    const float bo = bias[3 * H_ + ng];

    float creg[8];                                    // c-state in VGPRs
#pragma unroll
    for (int r = 0; r < 8; ++r) creg[r] = 0.0f;

    const __bf16* axrow = xbf + (size_t)m * (S_ * F_) + hi * 8;  // + t*F_ per step

    auto load_b = [&](int g, int kt) -> bf16x16 {
        return *(const bf16x16*)(bsh + ((size_t)(g * NKT + kt) * 32 + lane) * 16);
    };

    for (int t = 0; t < S_; ++t) {
        const __bf16* hp = (t & 1) ? h1 : h0;
        __bf16*       hn = (t & 1) ? h0 : h1;
        const __bf16* axbase = axrow + (size_t)t * F_;
        const __bf16* ahbase = hp + (size_t)m * H_ + hi * 8;

        // x-vs-h source is selected at COMPILE TIME (loop fully unrolled)
        auto load_a = [&](int kt) -> bf16x16 {
            const __bf16* p = (kt < NKTX) ? (axbase + kt * 32)
                                          : (ahbase + (kt - NKTX) * 32);
            bf16x8 lo  = *(const bf16x8*)p;
            bf16x8 hi8 = *(const bf16x8*)(p + 16);
            return __builtin_shufflevector(lo, hi8,
                       0,1,2,3,4,5,6,7,8,9,10,11,12,13,14,15);
        };

        f32x8 acc0 = {}, acc1 = {}, acc2 = {}, acc3 = {};

        // software-pipelined, fully-unrolled K loop
        bf16x16 aC  = load_a(0);
        bf16x16 b0C = load_b(0, 0);
        bf16x16 b1C = load_b(1, 0);
        bf16x16 b2C = load_b(2, 0);
        bf16x16 b3C = load_b(3, 0);
#pragma unroll
        for (int kt = 0; kt < NKT; ++kt) {
            bf16x16 aN, b0N, b1N, b2N, b3N;
            if (kt + 1 < NKT) {
                aN  = load_a(kt + 1);
                b0N = load_b(0, kt + 1);
                b1N = load_b(1, kt + 1);
                b2N = load_b(2, kt + 1);
                b3N = load_b(3, kt + 1);
            }
            acc0 = __builtin_amdgcn_wmma_f32_16x16x32_bf16(false, aC, false, b0C, (short)0, acc0, false, false);
            acc1 = __builtin_amdgcn_wmma_f32_16x16x32_bf16(false, aC, false, b1C, (short)0, acc1, false, false);
            acc2 = __builtin_amdgcn_wmma_f32_16x16x32_bf16(false, aC, false, b2C, (short)0, acc2, false, false);
            acc3 = __builtin_amdgcn_wmma_f32_16x16x32_bf16(false, aC, false, b3C, (short)0, acc3, false, false);
            if (kt + 1 < NKT) {
                aC = aN; b0C = b0N; b1C = b1N; b2C = b2N; b3C = b3N;
            }
        }

        // fused LSTM elementwise (reference: o is NOT sigmoided)
#pragma unroll
        for (int r = 0; r < 8; ++r) {
            const int brow = mt * 16 + r + hi * 8;    // C/D row -> batch index
            float vi = sigmoid_fast(acc0[r] + bi);
            float vf = sigmoid_fast(acc1[r] + bf);
            float vg = tanhf(acc2[r] + bg);
            float vo = acc3[r] + bo;
            float c  = vf * creg[r] + vi * vg;
            creg[r] = c;
            float h = vo * tanhf(c) + vo;
            out[((size_t)brow * S_ + t) * H_ + ng] = h;
            hn[(size_t)brow * H_ + ng] = (__bf16)h;
        }

        // ---- software grid barrier between timesteps ----
        __threadfence();                              // release my h stores
        __syncthreads();
        if (threadIdx.x == 0) {
            atomicAdd(barrier, 1u);
            const unsigned tgt = (unsigned)NBLK * (unsigned)(t + 1);
            while (__hip_atomic_load(barrier, __ATOMIC_ACQUIRE,
                                     __HIP_MEMORY_SCOPE_AGENT) < tgt) {
                __builtin_amdgcn_s_sleep(2);
            }
        }
        __syncthreads();
        __threadfence();                              // acquire fresh h
    }
}

// ---------------------------------------------------------------------------
// Workspace layout (bytes):
//   [0, 12582912)            wfrag : bf16 swizzled [W_ih;W_hh] (12 MB)
//   [12582912, 79691776)     xbf   : bf16 x (64 MB)
//   [79691776, 79953920)     h ping
//   [79953920, 80216064)     h pong
//   [80216064, +4)           grid barrier counter
// ---------------------------------------------------------------------------
extern "C" void kernel_launch(void* const* d_in, const int* in_sizes, int n_in,
                              void* d_out, int out_size, void* d_ws, size_t ws_size,
                              hipStream_t stream) {
    const float* x    = (const float*)d_in[0];
    const float* w_ih = (const float*)d_in[1];
    const float* w_hh = (const float*)d_in[2];
    const float* bias = (const float*)d_in[3];
    float* out = (float*)d_out;

    char* ws = (char*)d_ws;
    __bf16*   wfrag = (__bf16*)(ws);
    __bf16*   xbf   = (__bf16*)(ws + 12582912);
    __bf16*   h0    = (__bf16*)(ws + 79691776);
    __bf16*   h1    = (__bf16*)(ws + 79953920);
    unsigned* bar   = (unsigned*)(ws + 80216064);

    pack_w_kernel<<<(G4 * KTOT) / 256, 256, 0, stream>>>(w_ih, w_hh, wfrag);
    pack_x_kernel<<<(B_ * S_ * F_) / 256, 256, 0, stream>>>(x, xbf, B_ * S_ * F_);
    init_state_kernel<<<(B_ * H_) / 256, 256, 0, stream>>>(h0, h1, bar);

    lstm_persistent_kernel<<<NBLK, 128, 0, stream>>>(
        xbf, wfrag, bias, h0, h1, out, bar);
}
